// IncrementalAttention_35699768165026
// MI455X (gfx1250) — matrix operand order repaired and checked by reading, not compiled, not run on hardware
//
#include <hip/hip_runtime.h>

#define B_   2
#define SQ_  2048
#define SE_  2048
#define SKV_ 4096
#define H_   1024
#define NH_  16
#define HD_  64

typedef __attribute__((ext_vector_type(16))) __bf16 bf16x16;
typedef __attribute__((ext_vector_type(8)))  float  f32x8;
typedef int v4i __attribute__((vector_size(16)));   // matches builtin 'V4i' param type

#define AS1 __attribute__((address_space(1)))
#define AS3 __attribute__((address_space(3)))

#if __has_builtin(__builtin_amdgcn_global_load_async_to_lds_b128) && \
    __has_builtin(__builtin_amdgcn_s_wait_asynccnt)
#define USE_ASYNC 1
#else
#define USE_ASYNC 0
#endif

// fp32 -> bf16 round-to-nearest-even
static __device__ __forceinline__ unsigned short f2bf(float f) {
    unsigned int u = __float_as_uint(f);
    unsigned int r = u + 0x7FFFu + ((u >> 16) & 1u);
    return (unsigned short)(r >> 16);
}

// Assemble a 16-element bf16 fragment from two 16-byte chunks (global or LDS).
static __device__ __forceinline__ bf16x16 ldpair(const unsigned short* p0,
                                                 const unsigned short* p1) {
    union { bf16x16 v; uint4 q[2]; } u;
    u.q[0] = *(const uint4*)p0;
    u.q[1] = *(const uint4*)p1;
    return u.v;
}

static __device__ __forceinline__ f32x8 wmma_bf16(bf16x16 a, bf16x16 b, f32x8 c) {
    return __builtin_amdgcn_wmma_f32_16x16x32_bf16(
        /*neg_a=*/false, a, /*neg_b=*/false, b,
        /*c_mod=*/(short)0, c, /*reuse_a=*/false, /*reuse_b=*/false);
}

// 16-byte global -> LDS copy: async path (ASYNCcnt) if available, sync fallback.
static __device__ __forceinline__ void cp16(const unsigned short* g, unsigned short* l) {
#if USE_ASYNC
    __builtin_amdgcn_global_load_async_to_lds_b128(
        (AS1 v4i*)(void*)g,   // generic -> addrspace(1) cast (drops const)
        (AS3 v4i*)(void*)l,   // generic -> addrspace(3) cast
        0, 0);
#else
    *(uint4*)l = *(const uint4*)g;
#endif
}

// ---------------- conversion kernels ----------------

__global__ void attn_cvt_inputs(const float* __restrict__ hs,
                                const float* __restrict__ ehs,
                                unsigned short* __restrict__ X) {
    size_t i = (size_t)blockIdx.x * 256 + threadIdx.x;
    int h = (int)(i & (H_ - 1));
    size_t t = i >> 10;
    int s = (int)(t & (SKV_ - 1));
    int b = (int)(t >> 12);
    float v = (s < SQ_)
        ? hs[((size_t)b * SQ_ + s) * H_ + h]
        : ehs[((size_t)b * SE_ + (s - SQ_)) * H_ + h];
    X[i] = f2bf(v);
}

__global__ void attn_cvt_weights(const float* __restrict__ Wq,
                                 const float* __restrict__ Wk,
                                 const float* __restrict__ Wv,
                                 unsigned short* __restrict__ WT) {
    size_t i = (size_t)blockIdx.x * 256 + threadIdx.x;  // 3 * H * H
    int which = (int)(i >> 20);
    int r = (int)(i & ((1u << 20) - 1));
    int n = r >> 10, k = r & (H_ - 1);
    const float* W = (which == 0) ? Wq : (which == 1) ? Wk : Wv;
    WT[i] = f2bf(W[(size_t)k * H_ + n]);
}

// ---------------- QKV projection GEMM (bf16 WMMA, f32 accum) ----------------
__global__ __launch_bounds__(128)
void attn_qkv_proj(const unsigned short* __restrict__ Xbf,
                   const unsigned short* __restrict__ Wt,
                   const float* __restrict__ bias,
                   unsigned short* __restrict__ outp,
                   int mode) {
    int wave = threadIdx.x >> 5;
    int lane = threadIdx.x & 31;
    int lanelo = lane & 15;
    int hi = lane >> 4;

    int tile = blockIdx.x * 4 + wave;
    int tm = tile >> 6;
    int tn = tile & 63;

    int row  = tm * 16 + lanelo;
    int xrow = (mode == 0) ? ((row >> 11) * SKV_ + (row & 2047)) : row;
    const unsigned short* xr = Xbf + (size_t)xrow * H_;

    int cn = tn * 16 + lanelo;
    const unsigned short* wr = Wt + (size_t)cn * H_;

    f32x8 c;
    float bv = bias[cn];
#pragma unroll
    for (int j = 0; j < 8; j++) c[j] = bv;

    int ca = hi * 8;
    int cb = hi * 8;
#pragma unroll 4
    for (int k0 = 0; k0 < H_; k0 += 32) {
        bf16x16 a = ldpair(xr + k0 + ca, xr + k0 + ca + 16);
        bf16x16 w = ldpair(wr + k0 + cb, wr + k0 + cb + 16);
        c = wmma_bf16(a, w, c);
    }

    int head = cn >> 6, hd = cn & 63;
#pragma unroll
    for (int j = 0; j < 8; j++) {
        int rm = tm * 16 + j + hi * 8;
        unsigned short v = f2bf(c[j]);
        if (mode == 0) {
            int b = rm >> 11, sq = rm & 2047;
            outp[(((size_t)b * NH_ + head) * SQ_ + sq) * HD_ + hd] = v;
        } else if (mode == 1) {
            int b = rm >> 12, skv = rm & 4095;
            outp[(((size_t)b * NH_ + head) * SKV_ + skv) * HD_ + hd] = v;
        } else {
            int b = rm >> 12, skv = rm & 4095;
            outp[(((size_t)b * NH_ + head) * HD_ + hd) * SKV_ + skv] = v;
        }
    }
}

// ---------------- flash attention (bf16 WMMA, async LDS staging) ----------------
// 4 waves/block share one (b,head); K/V chunks double-buffered in LDS.
__global__ __launch_bounds__(128)
void attn_core(const unsigned short* __restrict__ Qbf,
               const unsigned short* __restrict__ Kbf,
               const unsigned short* __restrict__ VbfT,
               const float* __restrict__ mask,
               float* __restrict__ out) {
    __shared__ __align__(16) unsigned short Kt[2][32][64];   // [buf][kv][hd]
    __shared__ __align__(16) unsigned short Vt[2][64][32];   // [buf][hd][kv]
    __shared__ __align__(16) unsigned short pbuf[4][16 * 32];

    int wave = threadIdx.x >> 5;
    int lane = threadIdx.x & 31;
    int lanelo = lane & 15;
    int hi = lane >> 4;
    int hi8 = hi * 8;
    int tid = threadIdx.x;

    int bh = blockIdx.x >> 5;
    int qt = (blockIdx.x & 31) * 4 + wave;    // q tile 0..127
    int b = bh / NH_;
    int head = bh % NH_;

    const unsigned short* qr =
        Qbf + ((size_t)bh * SQ_ + qt * 16 + lanelo) * HD_;
    int ca = hi * 8;
    bf16x16 aq0 = ldpair(qr + ca,      qr + ca + 16);
    bf16x16 aq1 = ldpair(qr + 32 + ca, qr + 32 + ca + 16);

    const unsigned short* Kp = Kbf  + (size_t)bh * SKV_ * HD_;
    const unsigned short* Vp = VbfT + (size_t)bh * HD_ * SKV_;
    const float* mrow = mask + (size_t)b * SKV_;

    // stage one 32-kv chunk of K and V into LDS (4 x 16B per thread)
    auto stage = [&](int buf, int kv0) {
#pragma unroll
        for (int r = 0; r < 2; r++) {
            int c = tid + r * 128;             // 0..255: K tile, 32 rows x 8 chunks
            int row = c >> 3, sub = (c & 7) * 8;
            cp16(Kp + (size_t)(kv0 + row) * HD_ + sub, &Kt[buf][row][sub]);
        }
#pragma unroll
        for (int r = 0; r < 2; r++) {
            int c = tid + r * 128;             // 0..255: V tile, 64 rows x 4 chunks
            int row = c >> 2, sub = (c & 3) * 8;
            cp16(Vp + (size_t)row * SKV_ + kv0 + sub, &Vt[buf][row][sub]);
        }
    };

    f32x8 o[4] = {};
    float mj[8], lj[8];
#pragma unroll
    for (int j = 0; j < 8; j++) { mj[j] = -3.0e38f; lj[j] = 0.0f; }

    stage(0, 0);
    for (int kv0 = 0; kv0 < SKV_; kv0 += 32) {
        int cur = (kv0 >> 5) & 1;
        bool havenext = (kv0 + 32) < SKV_;
        if (havenext) stage(cur ^ 1, kv0 + 32);   // prefetch next chunk
#if USE_ASYNC
        if (havenext) __builtin_amdgcn_s_wait_asynccnt(4);  // current chunk landed
        else          __builtin_amdgcn_s_wait_asynccnt(0);
#endif
        __syncthreads();   // barrier A: Kt/Vt[cur] visible to all waves

        // scores: two 16-col kv tiles, K-dim = HD (2 chained WMMAs each)
        f32x8 s0 = {}, s1 = {};
        {
            const unsigned short* kr = &Kt[cur][lanelo][hi * 8];
            bf16x16 kb0 = ldpair(kr,      kr + 16);
            bf16x16 kb1 = ldpair(kr + 32, kr + 48);
            s0 = wmma_bf16(aq0, kb0, s0);
            s0 = wmma_bf16(aq1, kb1, s0);
        }
        {
            const unsigned short* kr = &Kt[cur][16 + lanelo][hi * 8];
            bf16x16 kb0 = ldpair(kr,      kr + 16);
            bf16x16 kb1 = ldpair(kr + 32, kr + 48);
            s1 = wmma_bf16(aq0, kb0, s1);
            s1 = wmma_bf16(aq1, kb1, s1);
        }

        float mv0 = mrow[kv0 + lanelo];
        float mv1 = mrow[kv0 + 16 + lanelo];

        // online softmax; rows live across the 16 lanes of each half-wave
#pragma unroll
        for (int j = 0; j < 8; j++) {
            float a  = s0[j] * 0.125f + mv0;
            float bb = s1[j] * 0.125f + mv1;
            float rv = fmaxf(a, bb);
#pragma unroll
            for (int off = 1; off < 16; off <<= 1)
                rv = fmaxf(rv, __shfl_xor(rv, off, 32));
            float nm  = fmaxf(mj[j], rv);
            float fac = __expf(mj[j] - nm);
            mj[j] = nm;
            float p0 = __expf(a - nm);
            float p1 = __expf(bb - nm);
            float rs = p0 + p1;
#pragma unroll
            for (int off = 1; off < 16; off <<= 1)
                rs += __shfl_xor(rs, off, 32);
            lj[j] = lj[j] * fac + rs;
            o[0][j] *= fac; o[1][j] *= fac; o[2][j] *= fac; o[3][j] *= fac;
            pbuf[wave][(j + hi8) * 32 + lanelo]      = f2bf(p0);
            pbuf[wave][(j + hi8) * 32 + 16 + lanelo] = f2bf(p1);
        }
        // pbuf is private per wave; LDS ops from one wave are in-order,
        // so no block barrier is needed before re-reading it in A-layout.
        const unsigned short* pr = &pbuf[wave][lanelo * 32 + ca];
        bf16x16 ap = ldpair(pr, pr + 16);

        // ctx += P @ V : 4 hd tiles of 16, B-frags from LDS (contiguous kv)
#pragma unroll
        for (int h4 = 0; h4 < 4; h4++) {
            const unsigned short* vr = &Vt[cur][h4 * 16 + lanelo][hi * 8];
            bf16x16 vb = ldpair(vr, vr + 16);
            o[h4] = wmma_bf16(ap, vb, o[h4]);
        }
        __syncthreads();   // barrier C: done reading cur before it is re-staged
    }

    // epilogue: divide by softmax denominator, write f32 [B, SQ, H]
#pragma unroll
    for (int j = 0; j < 8; j++) {
        float inv = 1.0f / lj[j];
        int sq = qt * 16 + j + hi8;
        size_t base = ((size_t)b * SQ_ + sq) * H_ + head * HD_;
#pragma unroll
        for (int h4 = 0; h4 < 4; h4++)
            out[base + h4 * 16 + lanelo] = o[h4][j] * inv;
    }
}

// ---------------- launcher ----------------
extern "C" void kernel_launch(void* const* d_in, const int* in_sizes, int n_in,
                              void* d_out, int out_size, void* d_ws, size_t ws_size,
                              hipStream_t stream) {
    const float* hs   = (const float*)d_in[0];
    const float* ehs  = (const float*)d_in[1];
    const float* mask = (const float*)d_in[2];
    const float* Wq   = (const float*)d_in[3];
    const float* bq   = (const float*)d_in[4];
    const float* Wk   = (const float*)d_in[5];
    const float* bk   = (const float*)d_in[6];
    const float* Wv   = (const float*)d_in[7];
    const float* bv   = (const float*)d_in[8];

    unsigned short* ws = (unsigned short*)d_ws;
    unsigned short* Xbf = ws;                        // 8,388,608 elems
    unsigned short* WT  = ws + (size_t)8388608;      // 3,145,728 (Wq^T,Wk^T,Wv^T)
    unsigned short* Qbf = ws + (size_t)11534336;     // 4,194,304
    unsigned short* Kbf = ws + (size_t)15728640;     // 8,388,608
    unsigned short* Vbf = ws + (size_t)24117248;     // 8,388,608 (transposed)

    attn_cvt_inputs<<<32768, 256, 0, stream>>>(hs, ehs, Xbf);
    attn_cvt_weights<<<12288, 256, 0, stream>>>(Wq, Wk, Wv, WT);

    attn_qkv_proj<<<4096, 128, 0, stream>>>(Xbf, WT,               bq, Qbf, 0);
    attn_qkv_proj<<<8192, 128, 0, stream>>>(Xbf, WT + 1048576,     bk, Kbf, 1);
    attn_qkv_proj<<<8192, 128, 0, stream>>>(Xbf, WT + 2 * 1048576, bv, Vbf, 2);

    attn_core<<<1024, 128, 0, stream>>>(Qbf, Kbf, Vbf, mask, (float*)d_out);
}
